// TernaryPackedLinear_5918464934122
// MI455X (gfx1250) — matrix operand order repaired and checked by reading, not compiled, not run on hardware
//
#include <hip/hip_runtime.h>

// CDNA5 / gfx1250 ternary-packed linear: y = (x @ W^T)*alpha + bias
// W dequantized from bit-packs via an LDS nibble-pair LUT into exact f16
// {-1,0,+1}; GEMM with v_wmma_f32_16x16x32_f16 (f32 accumulate).
// Ping-pong LDS pipeline: next tile's global loads are issued before the
// WMMA block so HBM latency hides under the matrix pipe; one barrier/iter.

typedef __attribute__((ext_vector_type(16))) _Float16 v16h;
typedef __attribute__((ext_vector_type(2)))  _Float16 v2h;
typedef __attribute__((ext_vector_type(8)))  float    v8f;

#define M_DIM 8192
#define K_DIM 4096
#define O_DIM 4096
#define NWORDS 128

#define BM 256
#define BN 128
#define BK 64
#define LDA 72   // BK + 8 halves pad -> 144B row stride, conflict-free b128 gathers
#define LDB 72

union FragU { v16h v; uint4 q[2]; };

__device__ __forceinline__ unsigned pk2h(float a, float b) {
    return __builtin_bit_cast(unsigned, __builtin_amdgcn_cvt_pkrtz(a, b));
}

__global__ __launch_bounds__(256)
void ternary_wmma_kernel(const float* __restrict__ x,
                         const int*   __restrict__ nz,
                         const int*   __restrict__ sgn,
                         const float* __restrict__ bias,
                         const float* __restrict__ alpha,
                         float*       __restrict__ out)
{
    __shared__ __align__(16) _Float16 As[2][BM * LDA];  // x tiles, f16, [m][k]
    __shared__ __align__(16) _Float16 Bs[2][BN * LDB];  // W tiles, f16, [o][k]
    __shared__ __align__(8)  uint2    Wlut[256];        // nibble-pair -> 4 f16 weights

    const int tid    = threadIdx.x;
    const int lane   = tid & 31;
    const int wid    = tid >> 5;            // 0..7
    const int wave_m = (wid & 3) * 64;      // 4 waves along M, 64 rows each
    const int wave_n = (wid >> 2) * 64;     // 2 waves along N(O), 64 cols each
    const int lane_lo = lane & 15;
    const int lane_hi = lane >> 4;

    const int blockM = blockIdx.y * BM;
    const int blockN = blockIdx.x * BN;

    // ---- build decode LUT: idx = nz[3:0] | sgn[3:0]<<4 -> 4 packed f16 ----
    {
        const unsigned idx = (unsigned)tid;   // 256 threads, 256 entries
        unsigned lo = 0, hi = 0;
        #pragma unroll
        for (int j = 0; j < 2; ++j) {
            unsigned b = (idx >> j) & 1u, s = (idx >> (4 + j)) & 1u;
            lo |= (b * (0x3C00u | (s << 15))) << (16 * j);
        }
        #pragma unroll
        for (int j = 2; j < 4; ++j) {
            unsigned b = (idx >> j) & 1u, s = (idx >> (4 + j)) & 1u;
            hi |= (b * (0x3C00u | (s << 15))) << (16 * (j - 2));
        }
        Wlut[idx] = make_uint2(lo, hi);
    }

    // A staging: 1 thread per row, 64 floats; B staging: 2 threads per o-row
    const int arow  = tid;                  // 0..255
    const int brow  = tid >> 1;             // 0..127  (o within tile)
    const int bword = tid & 1;              // which 32-bit word of the 64-K chunk

    const float* xrow = x + (size_t)(blockM + arow) * K_DIM;
    const int*   nzrow  = nz  + (size_t)(blockN + brow) * NWORDS;
    const int*   sgnrow = sgn + (size_t)(blockN + brow) * NWORDS;

    v8f acc[4][4];
    {
        v8f z = {};
        #pragma unroll
        for (int i = 0; i < 4; ++i)
            #pragma unroll
            for (int j = 0; j < 4; ++j) acc[i][j] = z;
    }

    // ---------- prologue: stage k0 = 0 into buffer 0 ----------
    {
        uint4* asrow = (uint4*)&As[0][arow * LDA];
        #pragma unroll
        for (int i = 0; i < 8; ++i) {
            float4 f0 = ((const float4*)xrow)[2 * i];
            float4 f1 = ((const float4*)xrow)[2 * i + 1];
            asrow[i] = make_uint4(pk2h(f0.x, f0.y), pk2h(f0.z, f0.w),
                                  pk2h(f1.x, f1.y), pk2h(f1.z, f1.w));
        }
        const unsigned nzw = (unsigned)nzrow[bword];
        const unsigned sgw = (unsigned)sgnrow[bword];
        unsigned pk[16];
        #pragma unroll
        for (int g = 0; g < 8; ++g) {
            unsigned idx = ((nzw >> (4 * g)) & 0xFu) | (((sgw >> (4 * g)) & 0xFu) << 4);
            uint2 e = Wlut[idx];
            pk[2 * g] = e.x; pk[2 * g + 1] = e.y;
        }
        uint4* bsp = (uint4*)&Bs[0][brow * LDB + bword * 32];
        #pragma unroll
        for (int q = 0; q < 4; ++q)
            bsp[q] = make_uint4(pk[4*q], pk[4*q+1], pk[4*q+2], pk[4*q+3]);
    }
    __syncthreads();

    for (int k0 = 0; k0 < K_DIM; k0 += BK) {
        const int  p        = (k0 >> 6) & 1;
        const bool has_next = (k0 + BK) < K_DIM;
        const _Float16* curA = As[p];
        const _Float16* curB = Bs[p];

        // ---------- issue next tile's global loads (latency hides under WMMA) ----------
        float4 fx[16];
        unsigned nzw = 0, sgw = 0;
        if (has_next) {
            const float4* xp4 = (const float4*)(xrow + k0 + BK);
            __builtin_prefetch(xrow + k0 + 2 * BK, 0, 1);   // global_prefetch_b8
            #pragma unroll
            for (int i = 0; i < 16; ++i) fx[i] = xp4[i];    // global_load_b128, no wait yet
            const int widx = ((k0 + BK) >> 5) + bword;
            nzw = (unsigned)nzrow[widx];
            sgw = (unsigned)sgnrow[widx];
        }

        // ---------- compute: 2 k-steps of 32, 16 WMMA tiles each ----------
        #pragma unroll
        for (int ks = 0; ks < 2; ++ks) {
            const int kk = ks * 32;
            FragU a[4], b[4];
            // A frag: lane(0-15)=row M, halves 0-7 = K kb..kb+7, 8-15 = K kb+16..kb+23
            #pragma unroll
            for (int mt = 0; mt < 4; ++mt) {
                const _Float16* ap = &curA[(wave_m + mt * 16 + lane_lo) * LDA + kk + lane_hi * 8];
                a[mt].q[0] = *(const uint4*)ap;         // ds_load_b128
                a[mt].q[1] = *(const uint4*)(ap + 16);  // ds_load_b128
            }
            // B frag: lane = column N (=o), halves 0-7 = K kb..kb+7, 8-15 = K kb+16..kb+23
            #pragma unroll
            for (int nt = 0; nt < 4; ++nt) {
                const _Float16* bp = &curB[(wave_n + nt * 16 + lane_lo) * LDB + kk + lane_hi * 8];
                b[nt].q[0] = *(const uint4*)bp;
                b[nt].q[1] = *(const uint4*)(bp + 16);
            }
            #pragma unroll
            for (int mt = 0; mt < 4; ++mt)
                #pragma unroll
                for (int nt = 0; nt < 4; ++nt)
                    acc[mt][nt] = __builtin_amdgcn_wmma_f32_16x16x32_f16(
                        false, a[mt].v, false, b[nt].v,
                        (short)0, acc[mt][nt], false, false);
        }

        // ---------- drain next tile into the other buffer ----------
        if (has_next) {
            uint4* asrow = (uint4*)&As[p ^ 1][arow * LDA];
            #pragma unroll
            for (int i = 0; i < 8; ++i) {
                float4 f0 = fx[2 * i];
                float4 f1 = fx[2 * i + 1];
                asrow[i] = make_uint4(pk2h(f0.x, f0.y), pk2h(f0.z, f0.w),
                                      pk2h(f1.x, f1.y), pk2h(f1.z, f1.w)); // ds_store_b128
            }
            unsigned pk[16];
            #pragma unroll
            for (int g = 0; g < 8; ++g) {
                unsigned idx = ((nzw >> (4 * g)) & 0xFu) | (((sgw >> (4 * g)) & 0xFu) << 4);
                uint2 e = Wlut[idx];                 // ds_load_b64 (LUT gather)
                pk[2 * g] = e.x; pk[2 * g + 1] = e.y;
            }
            uint4* bsp = (uint4*)&Bs[p ^ 1][brow * LDB + bword * 32];
            #pragma unroll
            for (int q = 0; q < 4; ++q)
                bsp[q] = make_uint4(pk[4*q], pk[4*q+1], pk[4*q+2], pk[4*q+3]);
        }

        __syncthreads();   // single barrier per iteration (ping-pong)
    }

    // ---------- epilogue: y = acc*alpha[n] + bias[n] ----------
    // C/D layout: lane(0-15)=N, lanes 16-31 shift M by 8; VGPR r = M row r (+8)
    #pragma unroll
    for (int nt = 0; nt < 4; ++nt) {
        const int n  = blockN + wave_n + nt * 16 + lane_lo;
        const float al = alpha[n];
        const float bi = bias[n];
        #pragma unroll
        for (int mt = 0; mt < 4; ++mt) {
            const int mbase = blockM + wave_m + mt * 16 + lane_hi * 8;
            #pragma unroll
            for (int r = 0; r < 8; ++r)
                out[(size_t)(mbase + r) * O_DIM + n] = acc[mt][nt][r] * al + bi;
        }
    }
}

extern "C" void kernel_launch(void* const* d_in, const int* in_sizes, int n_in,
                              void* d_out, int out_size, void* d_ws, size_t ws_size,
                              hipStream_t stream) {
    const float* x     = (const float*)d_in[0];
    const int*   nz    = (const int*)  d_in[1];
    const int*   sgn   = (const int*)  d_in[2];
    const float* bias  = (const float*)d_in[3];
    const float* alpha = (const float*)d_in[4];
    float* out = (float*)d_out;

    dim3 grid(O_DIM / BN, M_DIM / BM);   // 32 x 32 workgroups
    ternary_wmma_kernel<<<grid, 256, 0, stream>>>(x, nz, sgn, bias, alpha, out);
}
